// SlotAttention_53472342835636
// MI455X (gfx1250) — matrix operand order repaired and checked by reading, not compiled, not run on hardware
//
#include <hip/hip_runtime.h>
#include <cstdint>
#include <cstddef>

// ---------------- problem constants ----------------
#define B_    4
#define N_    65536
#define DIM_  128
#define NS_   8
#define ITERS_ 3
#define SCALE_ 0.08838834764831845f   // 128^-0.5
#define EPS_   1e-8f

// packed-weight fragment offsets (in 512-element fragments)
#define OF_FC1 0     // K=64  (2 kt) x N=32  (2 nt) = 4 frags
#define OF_FC6 4     // K=32  (1 kt) x N=32  (2 nt) = 2 frags
#define OF_FC7 6     // K=256 (8 kt) x N=128 (8 nt) = 64 frags
#define OF_KW  70    // K=128 (4 kt) x N=128 (8 nt) = 32 frags
#define OF_VW  102   // 32 frags
#define NFRAGS 134

// attention partition
#define ABLK   64          // blocks per batch
#define AWAVES 4           // waves per block
#define NPART  (ABLK*AWAVES)       // 256 partials per batch
#define PTS_PER_WAVE (N_/NPART)    // 256
#define PSTRIDE (16*128 + 16)      // floats per partial (upd + sums)

typedef __attribute__((ext_vector_type(16))) __bf16 v16bf;
typedef __attribute__((ext_vector_type(8)))  float  v8f;

static_assert(sizeof(v16bf) == 32, "v16bf must be 32 bytes");

union Frag16 { v16bf v; uint4 q[2]; };
union H8     { __bf16 h[8]; uint4 q; };

__device__ __forceinline__ v8f wmma_bf16(v16bf a, v16bf b, v8f c) {
  // D = A(16x32) * B(32x16) + C(16x16), f32 accumulate
  return __builtin_amdgcn_wmma_f32_16x16x32_bf16(false, a, false, b, (short)0, c,
                                                 false, false);
}

// K offset (within a 32-wide K tile) of element e of a 16-bit A fragment
__device__ __forceinline__ int a_kofs(int half, int e) {
  return half * 16 + e;
}

// A fragment: lane holds row M=lane%16; 16 contiguous elems at K=half*16 in the row
__device__ __forceinline__ v16bf ld_a_lds(const __bf16* row, int half) {
  Frag16 f;
  f.q[0] = *reinterpret_cast<const uint4*>(row + half * 16);
  f.q[1] = *reinterpret_cast<const uint4*>(row + half * 16 + 8);
  return f.v;
}

// B fragment pre-packed (or naturally contiguous): 16 contiguous bf16 per lane
__device__ __forceinline__ v16bf ld_b16(const __bf16* p) {
  Frag16 f;
  f.q[0] = *reinterpret_cast<const uint4*>(p);
  f.q[1] = *reinterpret_cast<const uint4*>(p + 8);
  return f.v;
}

// ---------------- weight packing (B-fragment layout, bf16) ----------------
// layout: frag f = kt*(Nn/16)+nt ; element index ((f*32 + lane)*16 + e)
// value = W[kt*32 + half*16 + e][nt*16 + lane%16]  (zero-padded K)
__global__ void pack_b_kernel(const float* __restrict__ W, int Kreal, int Nn,
                              int Kpad, __bf16* __restrict__ dst) {
  int idx = blockIdx.x * blockDim.x + threadIdx.x;
  int total = (Kpad / 32) * (Nn / 16) * 512;
  if (idx >= total) return;
  int e    = idx & 15;
  int lane = (idx >> 4) & 31;
  int f    = idx >> 9;
  int ntiles = Nn / 16;
  int nt = f % ntiles, kt = f / ntiles;
  int half = lane >> 4;
  int n = nt * 16 + (lane & 15);
  int K = kt * 32 + half * 16 + e;
  float val = (K < Kreal) ? W[(size_t)K * Nn + n] : 0.0f;
  dst[idx] = (__bf16)val;
}

// ---------------- slot init ----------------
__global__ void init_slots_kernel(const float* __restrict__ mu,
                                  const float* __restrict__ logsig,
                                  const float* __restrict__ noise,
                                  float* __restrict__ slots) {
  int i = blockIdx.x * blockDim.x + threadIdx.x;
  if (i >= B_ * NS_ * DIM_) return;
  int d = i & (DIM_ - 1);
  slots[i] = mu[d] + __expf(logsig[d]) * noise[i];
}

// ---------------- fused front-end: fc1/fc6/fc7/LN/k/v ----------------
// 2 waves per block, each wave owns a 16-row tile of the flattened [B*N] points.
#define FE_WAVES 2
__global__ __launch_bounds__(64) void frontend_kernel(
    const float* __restrict__ inputs, const float* __restrict__ embedded,
    const float* __restrict__ fc1_b, const float* __restrict__ fc6_b,
    const float* __restrict__ fc7_b, const float* __restrict__ k_b,
    const float* __restrict__ v_b,
    const float* __restrict__ ln_g, const float* __restrict__ ln_b,
    const __bf16* __restrict__ wpack,
    __bf16* __restrict__ kout,   // [B*N][128] row-major bf16
    __bf16* __restrict__ vT) {   // [128][B*N] bf16
  __shared__ __bf16 xA[FE_WAVES][16][256];   // 16 KB : A staging (x225 then LN'd x | k)
  __shared__ __bf16 h1s[FE_WAVES][16][32];   //  2 KB
  __shared__ float  xf[FE_WAVES][16][128];   // 16 KB : f32 D staging for LN

  const int tid = threadIdx.x;
  const int wid = tid >> 5, lane = tid & 31, half = lane >> 4, l16 = lane & 15;
  const long row0 = (long)(blockIdx.x * FE_WAVES + wid) * 16;

  // stage inputs(0..63) | embedded(64..192) | zeros(193..255)
  for (int idx = lane; idx < 16 * 256; idx += 32) {
    int r = idx >> 8, c = idx & 255;
    long gr = row0 + r;
    float v = 0.f;
    if (c < 64)       v = inputs[gr * 64 + c];
    else if (c < 193) v = embedded[gr * 129 + (c - 64)];
    xA[wid][r][c] = (__bf16)v;
  }
  __syncthreads();

  // fc1: h1 = relu(inputs @ fc1 + b)   (K=64, N=32)
  for (int nt = 0; nt < 2; ++nt) {
    v8f acc = {};
    for (int kt = 0; kt < 2; ++kt) {
      v16bf a = ld_a_lds(&xA[wid][l16][kt * 32], half);
      v16bf b = ld_b16(wpack + ((size_t)(OF_FC1 + kt * 2 + nt) * 32 + lane) * 16);
      acc = wmma_bf16(a, b, acc);
    }
    float bias = fc1_b[nt * 16 + l16];
    for (int r = 0; r < 8; ++r) {
      float vv = acc[r] + bias;
      h1s[wid][r + half * 8][nt * 16 + l16] = (__bf16)(vv > 0.f ? vv : 0.f);
    }
  }
  __syncthreads();

  // fc6: h2 = h1 @ fc6 + b -> x225 cols 193..224   (K=32, N=32)
  for (int nt = 0; nt < 2; ++nt) {
    v8f acc = {};
    v16bf a = ld_a_lds(&h1s[wid][l16][0], half);
    v16bf b = ld_b16(wpack + ((size_t)(OF_FC6 + nt) * 32 + lane) * 16);
    acc = wmma_bf16(a, b, acc);
    float bias = fc6_b[nt * 16 + l16];
    for (int r = 0; r < 8; ++r)
      xA[wid][r + half * 8][193 + nt * 16 + l16] = (__bf16)(acc[r] + bias);
  }
  __syncthreads();

  // fc7: x = leaky_relu(x225 @ fc7 + b)   (K=256 padded, N=128)
  for (int nt = 0; nt < 8; ++nt) {
    v8f acc = {};
    for (int kt = 0; kt < 8; ++kt) {
      v16bf a = ld_a_lds(&xA[wid][l16][kt * 32], half);
      v16bf b = ld_b16(wpack + ((size_t)(OF_FC7 + kt * 8 + nt) * 32 + lane) * 16);
      acc = wmma_bf16(a, b, acc);
    }
    float bias = fc7_b[nt * 16 + l16];
    for (int r = 0; r < 8; ++r) {
      float vv = acc[r] + bias;
      xf[wid][r + half * 8][nt * 16 + l16] = (vv > 0.f) ? vv : 0.01f * vv;
    }
  }
  __syncthreads();

  // LayerNorm per row, write bf16 back into xA cols 0..127
  if (half == 0) {
    float m = 0.f;
    for (int c = 0; c < 128; ++c) m += xf[wid][l16][c];
    m *= (1.f / 128.f);
    float var = 0.f;
    for (int c = 0; c < 128; ++c) { float d = xf[wid][l16][c] - m; var += d * d; }
    var *= (1.f / 128.f);
    float rs = rsqrtf(var + 1e-5f);
    for (int c = 0; c < 128; ++c) {
      float xn = (xf[wid][l16][c] - m) * rs * ln_g[c] + ln_b[c];
      xA[wid][l16][c] = (__bf16)xn;
    }
  }
  __syncthreads();

  // k = x @ k_w + k_b -> stage bf16 into xA cols 128..255, then coalesced store
  for (int nt = 0; nt < 8; ++nt) {
    v8f acc = {};
    for (int kt = 0; kt < 4; ++kt) {
      v16bf a = ld_a_lds(&xA[wid][l16][kt * 32], half);
      v16bf b = ld_b16(wpack + ((size_t)(OF_KW + kt * 8 + nt) * 32 + lane) * 16);
      acc = wmma_bf16(a, b, acc);
    }
    float bias = k_b[nt * 16 + l16];
    for (int r = 0; r < 8; ++r)
      xA[wid][r + half * 8][128 + nt * 16 + l16] = (__bf16)(acc[r] + bias);
  }
  {
    int r = lane >> 1, part = lane & 1;   // lane -> (row, half-row)
    const uint4* src = reinterpret_cast<const uint4*>(&xA[wid][r][128 + part * 64]);
    uint4* dst = reinterpret_cast<uint4*>(kout + (size_t)(row0 + r) * 128 + part * 64);
    for (int i = 0; i < 8; ++i) dst[i] = src[i];
  }

  // v = x @ v_w + v_b -> store transposed directly (8 contiguous points per lane)
  const size_t BN = (size_t)B_ * N_;
  for (int nt = 0; nt < 8; ++nt) {
    v8f acc = {};
    for (int kt = 0; kt < 4; ++kt) {
      v16bf a = ld_a_lds(&xA[wid][l16][kt * 32], half);
      v16bf b = ld_b16(wpack + ((size_t)(OF_VW + kt * 8 + nt) * 32 + lane) * 16);
      acc = wmma_bf16(a, b, acc);
    }
    float bias = v_b[nt * 16 + l16];
    H8 t;
    for (int r = 0; r < 8; ++r) t.h[r] = (__bf16)(acc[r] + bias);
    int n = nt * 16 + l16;
    *reinterpret_cast<uint4*>(vT + (size_t)n * BN + row0 + half * 8) = t.q;
  }
}

// ---------------- slot pre: s=LN(slots), q=s@q_w+b, pack q A-fragments ----------------
__global__ __launch_bounds__(128) void slot_pre_kernel(
    const float* __restrict__ slots, const float* __restrict__ ln_g,
    const float* __restrict__ ln_b, const float* __restrict__ q_w,
    const float* __restrict__ q_b, __bf16* __restrict__ qpack) {
  __shared__ float sN[NS_][DIM_];
  __shared__ float q[NS_][DIM_];
  __shared__ float stats[NS_][2];
  const int b = blockIdx.x, tid = threadIdx.x;
  const float* S = slots + (size_t)b * NS_ * DIM_;
  if (tid < NS_) {
    float m = 0.f;
    for (int c = 0; c < DIM_; ++c) m += S[tid * DIM_ + c];
    m *= (1.f / DIM_);
    float v = 0.f;
    for (int c = 0; c < DIM_; ++c) { float d = S[tid * DIM_ + c] - m; v += d * d; }
    v *= (1.f / DIM_);
    stats[tid][0] = m;
    stats[tid][1] = rsqrtf(v + 1e-5f);
  }
  __syncthreads();
  for (int el = tid; el < NS_ * DIM_; el += 128) {
    int r = el >> 7, c = el & 127;
    sN[r][c] = (S[r * DIM_ + c] - stats[r][0]) * stats[r][1] * ln_g[c] + ln_b[c];
  }
  __syncthreads();
  for (int el = tid; el < NS_ * DIM_; el += 128) {
    int r = el >> 7, c = el & 127;
    float acc = q_b[c];
    for (int k = 0; k < DIM_; ++k) acc += sN[r][k] * q_w[k * DIM_ + c];
    q[r][c] = acc;
  }
  __syncthreads();
  {  // pack 4 A fragments (M=16 rows, rows 8..15 zero)
    int kt = tid >> 5, lane = tid & 31, half = lane >> 4, m = lane & 15;
    Frag16 f;
    __bf16* fp = reinterpret_cast<__bf16*>(&f);
    for (int e = 0; e < 16; ++e) {
      int K = kt * 32 + a_kofs(half, e);
      fp[e] = (m < NS_) ? (__bf16)q[m][K] : (__bf16)0.f;
    }
    uint4* dst = reinterpret_cast<uint4*>(qpack + ((size_t)(b * 4 + kt) * 32 + lane) * 16);
    dst[0] = f.q[0];
    dst[1] = f.q[1];
  }
}

// ---------------- attention: dots -> softmax(slots) -> accumulate updates ----------------
__global__ __launch_bounds__(128) void attn_kernel(
    const __bf16* __restrict__ kbf, const __bf16* __restrict__ vT,
    const __bf16* __restrict__ qpack, float* __restrict__ partials) {
  __shared__ __bf16 att[AWAVES][32][16];   // [point-in-tile][slot] staging for A frag
  __shared__ float  sred[AWAVES][16][8];
  const int b = blockIdx.y;
  const int tid = threadIdx.x, wid = tid >> 5, lane = tid & 31;
  const int half = lane >> 4, l16 = lane & 15;
  const int waveIdx = blockIdx.x * AWAVES + wid;
  const long pbase = (long)waveIdx * PTS_PER_WAVE;
  const size_t BN = (size_t)B_ * N_;
  const __bf16* kb = kbf + (size_t)b * N_ * DIM_;
  const __bf16* vb = vT + (size_t)b * N_;  // per-batch column offset in vT rows

  // zero the slot columns 8..15 once (pad rows of A)
  for (int i = lane; i < 32 * 16; i += 32) (&att[wid][0][0])[i] = (__bf16)0.f;

  v16bf qA[4];
  for (int kt = 0; kt < 4; ++kt)
    qA[kt] = ld_b16(qpack + ((size_t)(b * 4 + kt) * 32 + lane) * 16);

  v8f upd[8];
  for (int nt = 0; nt < 8; ++nt) upd[nt] = (v8f){};
  float sloc[8] = {0.f, 0.f, 0.f, 0.f, 0.f, 0.f, 0.f, 0.f};

  for (int t = 0; t < PTS_PER_WAVE; t += 32) {
    const long p0 = pbase + t;
    // dots for two 16-point subtiles; softmax over slots per point
    for (int sub = 0; sub < 2; ++sub) {
      const long ps = p0 + sub * 16;
      v8f dacc = {};
      for (int kt = 0; kt < 4; ++kt) {
        v16bf bf = ld_b16(kb + (size_t)(ps + l16) * DIM_ + kt * 32 + half * 16);
        dacc = wmma_bf16(qA[kt], bf, dacc);
      }
      if (half == 0) {  // lanes 0..15 hold slots 0..7 of their point
        float d[8], mx = -1e30f;
        for (int s = 0; s < 8; ++s) { d[s] = dacc[s] * SCALE_; mx = fmaxf(mx, d[s]); }
        float e[8], sum = 0.f;
        for (int s = 0; s < 8; ++s) { e[s] = __expf(d[s] - mx); sum += e[s]; }
        float inv = 1.f / sum;
        H8 tmp;
        for (int s = 0; s < 8; ++s) {
          float a = e[s] * inv + EPS_;
          sloc[s] += a;
          tmp.h[s] = (__bf16)a;
        }
        *reinterpret_cast<uint4*>(&att[wid][sub * 16 + l16][0]) = tmp.q;
      }
    }
    // build updates A fragment (M=slot, K=point)
    Frag16 af;
    __bf16* ap = reinterpret_cast<__bf16*>(&af);
    for (int e = 0; e < 16; ++e) ap[e] = att[wid][a_kofs(half, e)][l16];
    // accumulate updatesU += attnU(16x32) @ v(32x128)
    for (int nt = 0; nt < 8; ++nt) {
      v16bf bf = ld_b16(vb + (size_t)(nt * 16 + l16) * BN + p0 + half * 16);
      upd[nt] = wmma_bf16(af.v, bf, upd[nt]);
    }
  }

  // write per-wave partial (fixed order -> deterministic)
  float* P = partials + ((size_t)b * NPART + waveIdx) * PSTRIDE;
  for (int nt = 0; nt < 8; ++nt)
    for (int r = 0; r < 8; ++r)
      P[(r + half * 8) * 128 + nt * 16 + l16] = upd[nt][r];
  if (half == 0)
    for (int s = 0; s < 8; ++s) sred[wid][l16][s] = sloc[s];
  __syncthreads();
  if (lane < 8) {
    float s = 0.f;
    for (int i = 0; i < 16; ++i) s += sred[wid][i][lane];
    P[16 * 128 + lane] = s;
  }
}

// ---------------- reduce partials -> normalized updates ----------------
__global__ void reduce_kernel(const float* __restrict__ partials,
                              float* __restrict__ updates) {
  const int b = blockIdx.y;
  const int idx = blockIdx.x * blockDim.x + threadIdx.x;  // (s,d) in [0,1024)
  if (idx >= NS_ * DIM_) return;
  const int s = idx >> 7, d = idx & 127;
  const float* P = partials + (size_t)b * NPART * PSTRIDE;
  float acc = 0.f, ssum = 0.f;
  for (int i = 0; i < NPART; ++i) {
    acc  += P[(size_t)i * PSTRIDE + s * 128 + d];
    ssum += P[(size_t)i * PSTRIDE + 16 * 128 + s];
  }
  updates[((size_t)b * NS_ + s) * DIM_ + d] = acc / ssum;
}

// ---------------- slot post: GRU + residual MLP ----------------
__global__ __launch_bounds__(128) void slot_post_kernel(
    float* __restrict__ slots, const float* __restrict__ updates,
    const float* __restrict__ wih, const float* __restrict__ whh,
    const float* __restrict__ bih, const float* __restrict__ bhh,
    const float* __restrict__ w1, const float* __restrict__ b1,
    const float* __restrict__ w2, const float* __restrict__ b2,
    const float* __restrict__ ffg, const float* __restrict__ ffb,
    float* __restrict__ out, int write_out) {
  __shared__ float snew[NS_][DIM_];
  __shared__ float ffn[NS_][DIM_];
  __shared__ float hid[NS_][DIM_];
  __shared__ float stats[NS_][2];
  const int b = blockIdx.x, tid = threadIdx.x;
  float* S = slots + (size_t)b * NS_ * DIM_;
  const float* U = updates + (size_t)b * NS_ * DIM_;

  for (int el = tid; el < NS_ * DIM_; el += 128) {
    int s = el >> 7, d = el & 127;
    float xr = bih[d], xz = bih[128 + d], xn = bih[256 + d];
    float hr = bhh[d], hz = bhh[128 + d], hn = bhh[256 + d];
    for (int k = 0; k < DIM_; ++k) {
      float u = U[s * DIM_ + k], h = S[s * DIM_ + k];
      xr += u * wih[k * 384 + d];
      xz += u * wih[k * 384 + 128 + d];
      xn += u * wih[k * 384 + 256 + d];
      hr += h * whh[k * 384 + d];
      hz += h * whh[k * 384 + 128 + d];
      hn += h * whh[k * 384 + 256 + d];
    }
    float r = 1.f / (1.f + __expf(-(xr + hr)));
    float z = 1.f / (1.f + __expf(-(xz + hz)));
    float nn = tanhf(xn + r * hn);
    snew[s][d] = (1.f - z) * nn + z * S[s * DIM_ + d];
  }
  __syncthreads();
  if (tid < NS_) {
    float m = 0.f;
    for (int c = 0; c < DIM_; ++c) m += snew[tid][c];
    m *= (1.f / DIM_);
    float v = 0.f;
    for (int c = 0; c < DIM_; ++c) { float d = snew[tid][c] - m; v += d * d; }
    v *= (1.f / DIM_);
    stats[tid][0] = m;
    stats[tid][1] = rsqrtf(v + 1e-5f);
  }
  __syncthreads();
  for (int el = tid; el < NS_ * DIM_; el += 128) {
    int s = el >> 7, d = el & 127;
    ffn[s][d] = (snew[s][d] - stats[s][0]) * stats[s][1] * ffg[d] + ffb[d];
  }
  __syncthreads();
  for (int el = tid; el < NS_ * DIM_; el += 128) {
    int s = el >> 7, j = el & 127;
    float acc = b1[j];
    for (int k = 0; k < DIM_; ++k) acc += ffn[s][k] * w1[k * DIM_ + j];
    hid[s][j] = acc > 0.f ? acc : 0.f;
  }
  __syncthreads();
  for (int el = tid; el < NS_ * DIM_; el += 128) {
    int s = el >> 7, d = el & 127;
    float acc = b2[d];
    for (int j = 0; j < DIM_; ++j) acc += hid[s][j] * w2[j * DIM_ + d];
    float val = snew[s][d] + acc;
    S[s * DIM_ + d] = val;
    if (write_out) out[(size_t)b * NS_ * DIM_ + s * DIM_ + d] = val;
  }
}

// ---------------- host launch ----------------
extern "C" void kernel_launch(void* const* d_in, const int* in_sizes, int n_in,
                              void* d_out, int out_size, void* d_ws, size_t ws_size,
                              hipStream_t stream) {
  (void)in_sizes; (void)n_in; (void)out_size; (void)ws_size;
  const float* inputs   = (const float*)d_in[0];
  const float* embedded = (const float*)d_in[1];
  const float* noise    = (const float*)d_in[2];
  const float* mu       = (const float*)d_in[3];
  const float* logsig   = (const float*)d_in[4];
  const float* fc1_w = (const float*)d_in[5];
  const float* fc1_b = (const float*)d_in[6];
  const float* fc6_w = (const float*)d_in[7];
  const float* fc6_b = (const float*)d_in[8];
  const float* fc7_w = (const float*)d_in[9];
  const float* fc7_b = (const float*)d_in[10];
  const float* q_w = (const float*)d_in[11];
  const float* q_b = (const float*)d_in[12];
  const float* k_w = (const float*)d_in[13];
  const float* k_b = (const float*)d_in[14];
  const float* v_w = (const float*)d_in[15];
  const float* v_b = (const float*)d_in[16];
  const float* gwih = (const float*)d_in[17];
  const float* gwhh = (const float*)d_in[18];
  const float* gbih = (const float*)d_in[19];
  const float* gbhh = (const float*)d_in[20];
  const float* w1 = (const float*)d_in[21];
  const float* b1 = (const float*)d_in[22];
  const float* w2 = (const float*)d_in[23];
  const float* b2 = (const float*)d_in[24];
  const float* ln_in_g = (const float*)d_in[25];
  const float* ln_in_b = (const float*)d_in[26];
  const float* ln_sl_g = (const float*)d_in[27];
  const float* ln_sl_b = (const float*)d_in[28];
  const float* ln_ff_g = (const float*)d_in[29];
  const float* ln_ff_b = (const float*)d_in[30];

  // workspace layout
  uintptr_t base = (uintptr_t)d_ws;
  size_t off = 0;
  auto take = [&](size_t bytes) -> uintptr_t {
    uintptr_t p = base + off;
    off += (bytes + 255) & ~(size_t)255;
    return p;
  };
  __bf16* kbf     = (__bf16*)take((size_t)B_ * N_ * DIM_ * 2);     // 67 MB
  __bf16* vT      = (__bf16*)take((size_t)B_ * N_ * DIM_ * 2);     // 67 MB
  __bf16* wpack   = (__bf16*)take((size_t)NFRAGS * 512 * 2);
  __bf16* qpack   = (__bf16*)take((size_t)B_ * 4 * 512 * 2);
  float*  partials= (float*) take((size_t)B_ * NPART * PSTRIDE * 4);
  float*  updates = (float*) take((size_t)B_ * NS_ * DIM_ * 4);
  float*  slots   = (float*) take((size_t)B_ * NS_ * DIM_ * 4);
  float*  outp    = (float*)d_out;

  // pack weights into bf16 B-fragment layout
  pack_b_kernel<<<(4 * 512 + 255) / 256, 256, 0, stream>>>(fc1_w, 64, 32, 64, wpack + OF_FC1 * 512);
  pack_b_kernel<<<(2 * 512 + 255) / 256, 256, 0, stream>>>(fc6_w, 32, 32, 32, wpack + OF_FC6 * 512);
  pack_b_kernel<<<(64 * 512 + 255) / 256, 256, 0, stream>>>(fc7_w, 225, 128, 256, wpack + OF_FC7 * 512);
  pack_b_kernel<<<(32 * 512 + 255) / 256, 256, 0, stream>>>(k_w, 128, 128, 128, wpack + OF_KW * 512);
  pack_b_kernel<<<(32 * 512 + 255) / 256, 256, 0, stream>>>(v_w, 128, 128, 128, wpack + OF_VW * 512);

  init_slots_kernel<<<(B_ * NS_ * DIM_ + 255) / 256, 256, 0, stream>>>(mu, logsig, noise, slots);

  // fused front-end over all B*N points
  frontend_kernel<<<(B_ * N_) / (16 * FE_WAVES), 64, 0, stream>>>(
      inputs, embedded, fc1_b, fc6_b, fc7_b, k_b, v_b, ln_in_g, ln_in_b,
      wpack, kbf, vT);

  for (int it = 0; it < ITERS_; ++it) {
    slot_pre_kernel<<<B_, 128, 0, stream>>>(slots, ln_sl_g, ln_sl_b, q_w, q_b, qpack);
    attn_kernel<<<dim3(ABLK, B_), 128, 0, stream>>>(kbf, vT, qpack, partials);
    reduce_kernel<<<dim3(4, B_), 256, 0, stream>>>(partials, updates);
    slot_post_kernel<<<B_, 128, 0, stream>>>(slots, updates, gwih, gwhh, gbih, gbhh,
                                             w1, b1, w2, b2, ln_ff_g, ln_ff_b,
                                             outp, (it == ITERS_ - 1) ? 1 : 0);
  }
}
